// Net_4681514352669
// MI455X (gfx1250) — matrix operand order repaired and checked by reading, not compiled, not run on hardware
//
#include <hip/hip_runtime.h>
#include <hip/hip_fp16.h>
#include <string.h>
#include <stdint.h>

typedef _Float16 h16;
typedef __attribute__((ext_vector_type(16))) _Float16 v16h;
typedef __attribute__((ext_vector_type(8)))  _Float16 v8h;
typedef __attribute__((ext_vector_type(8)))  float    v8f;

#define LP 72  // LDS row pitch in halves (144B rows stay 16B-aligned, spreads banks)

// ---- CDNA5 async LDS<->global helpers (ASYNCcnt-tracked, no VGPR round-trip) ----
__device__ __forceinline__ void async_gl2lds_b128(uint32_t lds_off, const void* gaddr) {
  asm volatile("global_load_async_to_lds_b128 %0, %1, off"
               :: "v"(lds_off), "v"(gaddr) : "memory");
}
__device__ __forceinline__ void async_lds2gl_b128(void* gaddr, uint32_t lds_off) {
  asm volatile("global_store_async_from_lds_b128 %0, %1, off"
               :: "v"(gaddr), "v"(lds_off) : "memory");
}
__device__ __forceinline__ void wait_asynccnt0() {
  asm volatile("s_wait_asynccnt 0x0" ::: "memory");
}
__device__ __forceinline__ uint32_t lds_off32(const void* p) {
  // generic pointer to LDS: low 32 bits are the wave-relative LDS byte address
  return (uint32_t)(uintptr_t)p;
}

// ---------------- utility kernels ----------------
__global__ void zero_f32(float* p, int n) {
  int t = blockIdx.x * blockDim.x + threadIdx.x;
  if (t < n) p[t] = 0.f;
}

__global__ void copy_b16(const float4* __restrict__ s, float4* __restrict__ d, int n4) {
  int t = blockIdx.x * blockDim.x + threadIdx.x;
  if (t < n4) d[t] = s[t];
}

__global__ void f32_to_f16(const float* __restrict__ s, h16* __restrict__ d, int n) {
  int t = blockIdx.x * blockDim.x + threadIdx.x;
  if (t < n) d[t] = (h16)s[t];
}

// Pack a row-major (K x 64) fp32 weight into WMMA B-fragment order (f16).
// Fragment (kt,nt): lane L holds col nt*16+(L%16), K = kt*32 + (L/16)*8 + h (h<8), kt*32 + 16 + (L/16)*8 + (h-8) (h>=8),
// stored as 16 contiguous halves per lane (32B -> two b128 loads at use site).
__global__ void pack_w(const float* __restrict__ w, h16* __restrict__ out, int K) {
  int t = blockIdx.x * blockDim.x + threadIdx.x;
  if (t >= K * 64) return;
  int frag = t >> 9, within = t & 511;
  int lane = within >> 4, hh = within & 15;
  int kt = frag >> 2, nt = frag & 3;
  int k = kt * 32 + (lane >> 4) * 8 + (hh < 8 ? hh : 16 + (hh - 8));
  int n = nt * 16 + (lane & 15);
  out[t] = (h16)w[k * 64 + n];
}

// Scatter-add f16 rows of src into dst over batch-replicated edges.
// 8 lanes per edge, each gathers 16B (v8h) and issues 4 packed-f16 atomics
// (global_atomic_pk_add_f16): 2 channels per RMW -> half the atomic ops of f32.
__global__ void scatter_add_h(const h16* __restrict__ src, h16* dst,
                              const int* __restrict__ es, const int* __restrict__ ed,
                              int E, int ns, int nd, float* cnt) {
  int t = blockIdx.x * blockDim.x + threadIdx.x;
  if (t >= 32 * E * 8) return;
  int cb = t & 7;
  int eg = t >> 3;
  int b = eg / E, j = eg - b * E;
  int s = es[j] + b * ns;
  int d = ed[j] + b * nd;
  v8h v = *(const v8h*)(src + (size_t)s * 64 + (size_t)cb * 8);
  __half2 q[4];
  memcpy(q, &v, 16);
  __half2* o = (__half2*)(dst + (size_t)d * 64 + (size_t)cb * 8);
  unsafeAtomicAdd(o + 0, q[0]);
  unsafeAtomicAdd(o + 1, q[1]);
  unsafeAtomicAdd(o + 2, q[2]);
  unsafeAtomicAdd(o + 3, q[3]);
  if (cnt != nullptr && cb == 0) unsafeAtomicAdd(cnt + d, 1.0f);
}

// pooled = psum / max(cnt,1) (in place) and copy into z (GIN self term).
__global__ void pool_finalize_h(h16* psum, const float* __restrict__ cnt,
                                h16* __restrict__ z, int n) {
  int t = blockIdx.x * blockDim.x + threadIdx.x;
  if (t >= n) return;
  float v = (float)psum[t] / fmaxf(cnt[t >> 6], 1.0f);
  psum[t] = (h16)v;
  z[t] = (h16)v;
}

// Build the two K-half A fragments for a 16x64 f16 tile row, per CDNA5
// 16-bit A 16x32 layout: lane holds row lane%16, kbase=(lane/16)*8;
// halves[0..7]=K kbase..+7, halves[8..15]=K kbase+16..+23. p = row base + kbase.
__device__ __forceinline__ void build_a(const h16* p, v16h& alo, v16h& ahi) {
  v8h x0 = *(const v8h*)(p);
  v8h x1 = *(const v8h*)(p + 16);
  v8h x2 = *(const v8h*)(p + 32);
  v8h x3 = *(const v8h*)(p + 48);
#pragma unroll
  for (int j = 0; j < 8; ++j) {
    alo[j] = x0[j]; alo[8 + j] = x1[j];
    ahi[j] = x2[j]; ahi[8 + j] = x3[j];
  }
}

// Fused GIN MLP: out = relu( relu(bn(z@w1+b1)) @ w2 + b2 ), 64->64->64, f16 I/O.
// One 16-row tile per wave, 8 waves per block; out may alias z.
// Input tile staged with global_load_async_to_lds_b128; result drained with
// global_store_async_from_lds_b128 (ASYNCcnt), WMMA f16 with f32 accumulate.
__global__ void __launch_bounds__(256) gin_mlp(const h16* z, h16* out,
        const h16* __restrict__ w1f, const h16* __restrict__ w2f,
        const float* __restrict__ b1, const float* __restrict__ g,
        const float* __restrict__ bt, const float* __restrict__ rm,
        const float* __restrict__ rv, const float* __restrict__ b2) {
  __shared__ __align__(32) h16 lds[8][16 * LP];
  const int wave = threadIdx.x >> 5;
  const int lane = threadIdx.x & 31;
  h16* my = lds[wave];
  const size_t rowbase = ((size_t)blockIdx.x * 8 + wave) * 16;

  // stage z tile -> LDS directly (128 x 16B chunks, 4 per lane)
  for (int e = lane; e < 128; e += 32) {
    int r = e >> 3, c = (e & 7) * 8;
    async_gl2lds_b128(lds_off32(&my[r * LP + c]), z + (rowbase + r) * 64 + c);
  }
  wait_asynccnt0();
  __syncthreads();

  const int arow = lane & 15;
  const int kbase = (lane >> 4) * 8;
  const h16* ap = &my[arow * LP + kbase];
  v16h alo, ahi;
  build_a(ap, alo, ahi);

  v8f acc[4];
#pragma unroll
  for (int nt = 0; nt < 4; ++nt) {            // frag index = kt*4+nt
    v16h blo = *(const v16h*)(w1f + (size_t)(nt)     * 512 + (size_t)lane * 16);
    v16h bhi = *(const v16h*)(w1f + (size_t)(4 + nt) * 512 + (size_t)lane * 16);
    v8f c = {};
    c = __builtin_amdgcn_wmma_f32_16x16x32_f16(false, alo, false, blo, (short)0, c, false, false);
    c = __builtin_amdgcn_wmma_f32_16x16x32_f16(false, ahi, false, bhi, (short)0, c, false, false);
    acc[nt] = c;
  }
  __syncthreads();

  // BN-fold + ReLU, intermediate back to LDS (f16). C layout: N = lane%16 for
  // all 8 regs; M = v + 8*(lane/16).
#pragma unroll
  for (int nt = 0; nt < 4; ++nt) {
    int col = nt * 16 + (lane & 15);
    float sc = g[col] * rsqrtf(rv[col] + 1e-5f);
    float sh = (b1[col] - rm[col]) * sc + bt[col];
#pragma unroll
    for (int v = 0; v < 8; ++v) {
      float h = fmaxf(acc[nt][v] * sc + sh, 0.f);
      int rr = (lane >> 4) * 8 + v;
      my[rr * LP + col] = (h16)h;
    }
  }
  __syncthreads();

  build_a(ap, alo, ahi);
#pragma unroll
  for (int nt = 0; nt < 4; ++nt) {
    v16h blo = *(const v16h*)(w2f + (size_t)(nt)     * 512 + (size_t)lane * 16);
    v16h bhi = *(const v16h*)(w2f + (size_t)(4 + nt) * 512 + (size_t)lane * 16);
    v8f c = {};
    c = __builtin_amdgcn_wmma_f32_16x16x32_f16(false, alo, false, blo, (short)0, c, false, false);
    c = __builtin_amdgcn_wmma_f32_16x16x32_f16(false, ahi, false, bhi, (short)0, c, false, false);
    int col = nt * 16 + (lane & 15);
    float bb = b2[col];
#pragma unroll
    for (int v = 0; v < 8; ++v) {
      int rr = (lane >> 4) * 8 + v;
      my[rr * LP + col] = (h16)fmaxf(c[v] + bb, 0.f);
    }
  }
  __syncthreads();   // DS writes committed before async LDS reads below

  // drain the f16 result LDS -> global with async stores (out may alias z:
  // all reads of z completed before the first barrier)
  for (int e = lane; e < 128; e += 32) {
    int r = e >> 3, c = (e & 7) * 8;
    async_lds2gl_b128(out + (rowbase + r) * 64 + c, lds_off32(&my[r * LP + c]));
  }
  wait_asynccnt0();
}

// Readout lin1 with split-K: grid.x = 8 K-chunks of 32 kt each; 8 waves per
// block cover the 2x4 output tiles; partial sums accumulated into r1 with f32
// atomics (bias+ReLU folded into lin2_out after full accumulation).
__global__ void __launch_bounds__(256) readout_lin1(const h16* __restrict__ hf,
        const h16* __restrict__ wf, float* r1) {
  const int wave = threadIdx.x >> 5;
  const int lane = threadIdx.x & 31;
  const int mt = wave >> 2, nt = wave & 3;
  const int row = mt * 16 + (lane & 15);
  const int kb = (lane >> 4) * 8;
  const int kt0 = blockIdx.x * 32;
  v8f c = {};
  for (int kt = kt0; kt < kt0 + 32; ++kt) {
    const h16* p = hf + (size_t)row * 8192 + (size_t)kt * 32 + kb;
    v8h x0 = *(const v8h*)p;
    v8h x1 = *(const v8h*)(p + 16);
    v16h a;
#pragma unroll
    for (int j = 0; j < 8; ++j) { a[j] = x0[j]; a[8 + j] = x1[j]; }
    v16h b = *(const v16h*)(wf + ((size_t)kt * 4 + nt) * 512 + (size_t)lane * 16);
    c = __builtin_amdgcn_wmma_f32_16x16x32_f16(false, a, false, b, (short)0, c, false, false);
  }
  int col = nt * 16 + (lane & 15);
#pragma unroll
  for (int v = 0; v < 8; ++v) {
    int rr = mt * 16 + (lane >> 4) * 8 + v;
    unsafeAtomicAdd(r1 + rr * 64 + col, c[v]);
  }
}

// Final: out = relu(r1 + lin1_b) @ lin2_w + lin2_b. Tiny; plain VALU.
__global__ void lin2_out(const float* __restrict__ r1, const float* __restrict__ l1b,
                         const float* __restrict__ w, const float* __restrict__ b,
                         float* __restrict__ out) {
  int t = blockIdx.x * blockDim.x + threadIdx.x;
  if (t >= 320) return;
  int i = t / 10, j = t % 10;
  float s = b[j];
#pragma unroll
  for (int k = 0; k < 64; ++k)
    s += fmaxf(r1[i * 64 + k] + l1b[k], 0.f) * w[k * 10 + j];
  out[i * 10 + j] = s;
}

extern "C" void kernel_launch(void* const* d_in, const int* in_sizes, int n_in,
                              void* d_out, int out_size, void* d_ws, size_t ws_size,
                              hipStream_t stream) {
  (void)in_sizes; (void)n_in; (void)out_size; (void)ws_size;
  const int B = 32, N0 = 8192, N1 = 1024, N2 = 128;
  const int E0 = 65536, EC1 = 16384, EI1 = 16384, EC2 = 2048, EI2 = 2048;

  const float* x  = (const float*)d_in[0];
  const int* e0 = (const int*)d_in[2];
  const int* c1 = (const int*)d_in[3];
  const int* i1 = (const int*)d_in[4];
  const int* c2 = (const int*)d_in[5];
  const int* i2 = (const int*)d_in[6];
  const float* cv[8]; for (int k = 0; k < 8; ++k) cv[k] = (const float*)d_in[7 + k];
  const float* g1[8]; for (int k = 0; k < 8; ++k) g1[k] = (const float*)d_in[15 + k];
  const float* g2[8]; for (int k = 0; k < 8; ++k) g2[k] = (const float*)d_in[23 + k];
  const float* lin1_w = (const float*)d_in[31];
  const float* lin1_b = (const float*)d_in[32];
  const float* lin2_w = (const float*)d_in[33];
  const float* lin2_b = (const float*)d_in[34];
  float* out = (float*)d_out;

  // ---- workspace carve (~74 MB), all f16 activations ----
  char* base = (char*)d_ws;
  size_t off = 0;
  auto take = [&](size_t bytes) -> char* {
    char* p = base + off;
    off += (bytes + 255) & ~(size_t)255;
    return p;
  };
  h16* xh  = (h16*)take((size_t)B * N0 * 64 * 2);     // f16 copy of x
  h16* z0  = (h16*)take((size_t)B * N0 * 64 * 2);     // z/h level 0 (aliased)
  h16* p1  = (h16*)take((size_t)B * N1 * 64 * 2);     // pool sum / pooled 1
  float* n1c = (float*)take((size_t)B * N1 * 4);
  h16* z1  = (h16*)take((size_t)B * N1 * 64 * 2);
  h16* p2  = (h16*)take((size_t)B * N2 * 64 * 2);
  float* n2c = (float*)take((size_t)B * N2 * 4);
  h16* z2  = (h16*)take((size_t)B * N2 * 64 * 2);
  h16* pw[6]; for (int k = 0; k < 6; ++k) pw[k] = (h16*)take(64 * 64 * 2);
  h16* plin1 = (h16*)take((size_t)8192 * 64 * 2);
  float* r1  = (float*)take((size_t)B * 64 * 4);

  const int T = 256;
  auto blocks = [](long n, int t) { return (int)((n + t - 1) / t); };

  // pack all GEMM weights into WMMA B-fragment order (f16)
  pack_w<<<blocks(64 * 64, T), T, 0, stream>>>(cv[0], pw[0], 64);
  pack_w<<<blocks(64 * 64, T), T, 0, stream>>>(cv[6], pw[1], 64);
  pack_w<<<blocks(64 * 64, T), T, 0, stream>>>(g1[0], pw[2], 64);
  pack_w<<<blocks(64 * 64, T), T, 0, stream>>>(g1[6], pw[3], 64);
  pack_w<<<blocks(64 * 64, T), T, 0, stream>>>(g2[0], pw[4], 64);
  pack_w<<<blocks(64 * 64, T), T, 0, stream>>>(g2[6], pw[5], 64);
  pack_w<<<blocks((long)8192 * 64, T), T, 0, stream>>>(lin1_w, plin1, 8192);

  // level 0: xh = f16(x); z0 = xh + scatter_sum(xh); GIN MLP in place
  f32_to_f16<<<blocks((long)B * N0 * 64, T), T, 0, stream>>>(x, xh, B * N0 * 64);
  copy_b16<<<blocks((long)B * N0 * 8, T), T, 0, stream>>>((const float4*)xh, (float4*)z0, B * N0 * 8);
  scatter_add_h<<<blocks((long)B * E0 * 8, T), T, 0, stream>>>(xh, z0, e0, e0 + E0, E0, N0, N0, nullptr);
  gin_mlp<<<B * N0 / 128, T, 0, stream>>>(z0, z0, pw[0], pw[1], cv[1], cv[2], cv[3], cv[4], cv[5], cv[7]);

  // pool 0->1, level-1 GIN
  zero_f32<<<blocks((long)B * N1 * 32, T), T, 0, stream>>>((float*)p1, B * N1 * 32);
  zero_f32<<<blocks((long)B * N1, T), T, 0, stream>>>(n1c, B * N1);
  scatter_add_h<<<blocks((long)B * EC1 * 8, T), T, 0, stream>>>(z0, p1, c1, c1 + EC1, EC1, N0, N1, n1c);
  pool_finalize_h<<<blocks((long)B * N1 * 64, T), T, 0, stream>>>(p1, n1c, z1, B * N1 * 64);
  scatter_add_h<<<blocks((long)B * EI1 * 8, T), T, 0, stream>>>(p1, z1, i1, i1 + EI1, EI1, N1, N1, nullptr);
  gin_mlp<<<B * N1 / 128, T, 0, stream>>>(z1, z1, pw[2], pw[3], g1[1], g1[2], g1[3], g1[4], g1[5], g1[7]);

  // pool 1->2, level-2 GIN
  zero_f32<<<blocks((long)B * N2 * 32, T), T, 0, stream>>>((float*)p2, B * N2 * 32);
  zero_f32<<<blocks((long)B * N2, T), T, 0, stream>>>(n2c, B * N2);
  scatter_add_h<<<blocks((long)B * EC2 * 8, T), T, 0, stream>>>(z1, p2, c2, c2 + EC2, EC2, N1, N2, n2c);
  pool_finalize_h<<<blocks((long)B * N2 * 64, T), T, 0, stream>>>(p2, n2c, z2, B * N2 * 64);
  scatter_add_h<<<blocks((long)B * EI2 * 8, T), T, 0, stream>>>(p2, z2, i2, i2 + EI2, EI2, N2, N2, nullptr);
  gin_mlp<<<B * N2 / 128, T, 0, stream>>>(z2, z2, pw[4], pw[5], g2[1], g2[2], g2[3], g2[4], g2[5], g2[7]);

  // readout: z2 is already f16 row-major (32 x 8192); split-K over 8 WGPs
  zero_f32<<<blocks((long)B * 64, T), T, 0, stream>>>(r1, B * 64);
  readout_lin1<<<8, 256, 0, stream>>>(z2, plin1, r1);
  lin2_out<<<1, 320, 0, stream>>>(r1, lin1_b, lin2_w, lin2_b, out);
}
